// ABCLayer_28913719837435
// MI455X (gfx1250) — matrix operand order repaired and checked by reading, not compiled, not run on hardware
//
#include <hip/hip_runtime.h>
#include <hip/hip_fp16.h>

typedef __attribute__((ext_vector_type(16))) _Float16 v16h;
typedef __attribute__((ext_vector_type(4)))  _Float16 v4h;
typedef __attribute__((ext_vector_type(8)))  float    v8f;

#define HH     56
#define WW     56
#define CIN    128
#define COUT   128
#define BATCH  32
#define KW     147456            // 3*3*128*128
#define NPIX   (BATCH*HH*WW)     // 100352
#define MF     5

// LDS patch: 4 input rows x 58 cols x 128 ci (f16). Per-(row,col) position:
// 16 float4 data + 1 float4 pad -> 272B stride so A-fragment lanes (stride
// 1 col) rotate 4 banks/lane => conflict-free ds_load_b128.
#define PCOLS  58
#define PROWS  4
#define PSTRD  17
#define PSLOTS (PROWS * PCOLS * PSTRD)         // 3944 float4 = 63104 B

union U { float4 f4[2]; v16h h; };

__device__ __forceinline__ float sgnf(float x) {
    return (x > 0.f) ? 1.f : ((x < 0.f) ? -1.f : 0.f);
}

// ---------------- Kernel 1: mean / std of W -------------------------------
__global__ void k_stats(const float* __restrict__ Wsrc, float* __restrict__ scal) {
    __shared__ float s1[256], s2[256];
    float a = 0.f, b = 0.f;
    for (int k = threadIdx.x; k < KW; k += 256) {
        float w = Wsrc[k];
        a += w; b += w * w;
    }
    s1[threadIdx.x] = a; s2[threadIdx.x] = b;
    __syncthreads();
    for (int off = 128; off > 0; off >>= 1) {
        if ((int)threadIdx.x < off) {
            s1[threadIdx.x] += s1[threadIdx.x + off];
            s2[threadIdx.x] += s2[threadIdx.x + off];
        }
        __syncthreads();
    }
    if (threadIdx.x == 0) {
        float mean = s1[0] / (float)KW;
        float var  = s2[0] / (float)KW - mean * mean;
        scal[0] = mean;
        scal[1] = sqrtf(fmaxf(var, 0.f));
    }
}

// ---------------- Kernel 2: least-squares alpha (5x5 solve) ---------------
__global__ void k_alpha(const float* __restrict__ Wsrc,
                        const float* __restrict__ beta,
                        const float* __restrict__ bias,
                        int N, float* __restrict__ scal) {
    __shared__ float red[256];
    __shared__ float tot[20];
    const float mean = scal[0], stdv = scal[1];
    float u[MF];
#pragma unroll
    for (int i = 0; i < MF; i++) u[i] = -1.f + (float)i * (2.f / (MF - 1));

    float G[15], r[MF];
#pragma unroll
    for (int i = 0; i < 15; i++) G[i] = 0.f;
#pragma unroll
    for (int i = 0; i < MF; i++) r[i] = 0.f;

    for (int k = threadIdx.x; k < KW; k += 256) {
        float w = Wsrc[k];
        float bb[MF];
#pragma unroll
        for (int i = 0; i < MF; i++) bb[i] = sgnf(w - mean + u[i] * stdv);
        int q = 0;
#pragma unroll
        for (int i = 0; i < MF; i++) {
            r[i] += bb[i] * w;
#pragma unroll
            for (int j = i; j < MF; j++) G[q++] += bb[i] * bb[j];
        }
    }
    for (int q = 0; q < 20; q++) {
        red[threadIdx.x] = (q < 15) ? G[q] : r[q - 15];
        __syncthreads();
        for (int off = 128; off > 0; off >>= 1) {
            if ((int)threadIdx.x < off) red[threadIdx.x] += red[threadIdx.x + off];
            __syncthreads();
        }
        if (threadIdx.x == 0) tot[q] = red[0];
        __syncthreads();
    }
    if (threadIdx.x == 0) {
        float A[MF][MF], rhs[MF], alpha[MF];
        int q = 0;
        for (int i = 0; i < MF; i++)
            for (int j = i; j < MF; j++) { A[i][j] = tot[q]; A[j][i] = tot[q]; q++; }
        for (int i = 0; i < MF; i++) rhs[i] = tot[15 + i];
        for (int c = 0; c < MF; c++) {
            int piv = c; float best = fabsf(A[c][c]);
            for (int rr = c + 1; rr < MF; rr++) {
                float v2 = fabsf(A[rr][c]);
                if (v2 > best) { best = v2; piv = rr; }
            }
            if (piv != c) {
                for (int j = 0; j < MF; j++) { float t = A[c][j]; A[c][j] = A[piv][j]; A[piv][j] = t; }
                float t = rhs[c]; rhs[c] = rhs[piv]; rhs[piv] = t;
            }
            float inv = 1.f / A[c][c];
            for (int rr = c + 1; rr < MF; rr++) {
                float f = A[rr][c] * inv;
                for (int j = c; j < MF; j++) A[rr][j] -= f * A[c][j];
                rhs[rr] -= f * rhs[c];
            }
        }
        for (int i = MF - 1; i >= 0; i--) {
            float s = rhs[i];
            for (int j = i + 1; j < MF; j++) s -= A[i][j] * alpha[j];
            alpha[i] = s / A[i][i];
        }
        float sa = 0.f;
        for (int i = 0; i < MF; i++) { scal[2 + i] = alpha[i]; sa += alpha[i]; }
        scal[7] = sa;
        float bb = 0.f;
        for (int n = 0; n < N; n++) bb += beta[n] * bias[n];
        scal[8] = sa * bb;   // scalar added to every output element
    }
}

// ---------------- Kernel 3: W_eff in f16, packed [tap*4+kc][16h+n][cout tile][s] -----------------
__global__ void k_weff(const float* __restrict__ Wsrc,
                       const float* __restrict__ scal,
                       _Float16* __restrict__ Wh) {
    int idx = blockIdx.x * 256 + threadIdx.x;
    if (idx >= KW) return;
    float mean = scal[0], stdv = scal[1];
    float w = Wsrc[idx];
    float val = 0.f;
#pragma unroll
    for (int i = 0; i < MF; i++) {
        float u = -1.f + (float)i * (2.f / (MF - 1));
        val += scal[2 + i] * sgnf(w - mean + u * stdv);
    }
    Wh[((((idx >> 14) * 128 + (((idx >> 7) & 127) & 96) + ((((idx >> 7) & 31) >> 3) & 1) * 16 + (idx & 15)) * 128) + ((idx & 127) >> 4) * 16 + (((idx >> 7) & 7) + 8 * (((idx >> 7) & 31) >> 4))] = (_Float16)val;   // packed [tap*4+kc][16h+n][cout tile][s], k in WMMA table order
}

// ---------------- Kernel 4: combined ternary activation in f16 ------------
__global__ void k_act(const float* __restrict__ X,
                      const float* __restrict__ beta,
                      const float* __restrict__ vvec,
                      int N, _Float16* __restrict__ Ah, int n4) {
    int i4 = blockIdx.x * 256 + threadIdx.x;
    if (i4 >= n4) return;
    float4 x = reinterpret_cast<const float4*>(X)[i4];
    float o0 = 0.f, o1 = 0.f, o2 = 0.f, o3 = 0.f;
    for (int n = 0; n < N; n++) {
        float vn = vvec[n], bn = beta[n];
        o0 += bn * sgnf(fminf(fmaxf(x.x + vn, 0.f), 1.f) - 0.5f);
        o1 += bn * sgnf(fminf(fmaxf(x.y + vn, 0.f), 1.f) - 0.5f);
        o2 += bn * sgnf(fminf(fmaxf(x.z + vn, 0.f), 1.f) - 0.5f);
        o3 += bn * sgnf(fminf(fmaxf(x.w + vn, 0.f), 1.f) - 0.5f);
    }
    v4h r = { (_Float16)o0, (_Float16)o1, (_Float16)o2, (_Float16)o3 };
    reinterpret_cast<v4h*>(Ah)[i4] = r;
}

// One K=32 chunk: 8 WMMAs (2 output rows x 4 M-tiles) against one B fragment.
__device__ __forceinline__ void conv_chunk(int tk, const U& bu, v8f acc[8],
                                           const float4* sp, int mrow, int sel) {
    const int t  = tk >> 2;
    const int kc = tk & 3;
    const int dy = t / 3;
    const int dx = t - 3 * dy;
    const int fofs = kc * 4 + sel;
#pragma unroll
    for (int g = 0; g < 2; g++) {              // g = output row within pair
        U au0, au1, au2, au3;
#pragma unroll
        for (int mt = 0; mt < 4; mt++) {
            int col = mt * 16 + mrow + dx;
            if (col > PCOLS - 1) col = PCOLS - 1;   // over-pad; masked on store
            const int base = ((g + dy) * PCOLS + col) * PSTRD + fofs;
            U* au = (mt == 0) ? &au0 : (mt == 1) ? &au1 : (mt == 2) ? &au2 : &au3;
            au->f4[0] = sp[base];               // K 0..7   (upper half-wave: 8..15)
            au->f4[1] = sp[base + 2];           // K 16..23 (upper half-wave: 24..31)
        }
        acc[g * 4 + 0] = __builtin_amdgcn_wmma_f32_16x16x32_f16(
            false, au0.h, false, bu.h, (short)0, acc[g * 4 + 0], false, false);
        acc[g * 4 + 1] = __builtin_amdgcn_wmma_f32_16x16x32_f16(
            false, au1.h, false, bu.h, (short)0, acc[g * 4 + 1], false, false);
        acc[g * 4 + 2] = __builtin_amdgcn_wmma_f32_16x16x32_f16(
            false, au2.h, false, bu.h, (short)0, acc[g * 4 + 2], false, false);
        acc[g * 4 + 3] = __builtin_amdgcn_wmma_f32_16x16x32_f16(
            false, au3.h, false, bu.h, (short)0, acc[g * 4 + 3], false, false);
    }
}

// ---------------- Kernel 5: implicit-GEMM 3x3 conv via WMMA f16 -----------
// One block = one (image, row-pair). 256 threads = 8 waves; wave w owns cout
// tile w for BOTH rows (8 M-tiles of 16 pixels). K = 9 taps * 128 ci,
// flattened into 36 chunks of K=32, unrolled x2 with ping-pong B fragments
// (no register rotation, no WMMA->VALU WAR nops).
__global__ __launch_bounds__(256) void k_conv(const _Float16* __restrict__ Ah,
                                              const _Float16* __restrict__ Wh,
                                              const float* __restrict__ scal,
                                              float* __restrict__ out) {
    __shared__ float4 sp[PSLOTS];   // 63104 B

    const int tid = threadIdx.x;
    const int y0 = 2 * (blockIdx.x % (HH / 2));
    const int b  = blockIdx.x / (HH / 2);

    // Stage halo patch (rows y0-1..y0+2, cols -1..56) into LDS, 16B per iter.
    const float4* Aq = reinterpret_cast<const float4*>(Ah);
    for (int idx = tid; idx < PROWS * PCOLS * 16; idx += 256) {
        int p = idx >> 4, seg = idx & 15;
        int r = p / PCOLS, c = p % PCOLS;
        int gy = y0 + r - 1, gx = c - 1;
        float4 val = make_float4(0.f, 0.f, 0.f, 0.f);
        if (gy >= 0 && gy < HH && gx >= 0 && gx < WW)
            val = Aq[((size_t)(b * HH + gy) * WW + gx) * 16 + seg];
        sp[p * PSTRD + seg] = val;
    }
    __syncthreads();

    const int lane = tid & 31;
    const int wv   = tid >> 5;        // cout tile 0..7
    const int sel  = lane >> 4;       // half-wave K-group select for A fragment
    const int mrow = lane & 15;       // pixel row of A fragment carried by this lane

    const float4* Wq = reinterpret_cast<const float4*>(Wh);
    const int wfofs = wv * 2;

    v8f acc[8] = {v8f{}, v8f{}, v8f{}, v8f{}, v8f{}, v8f{}, v8f{}, v8f{}};

    // Ping-pong B fragments over 36 K-chunks, stepping 2 per trip.
    U b0, b1;
    b0.f4[0] = Wq[(0 * 32 + lane) * 16 + wfofs];
    b0.f4[1] = Wq[(0 * 32 + lane) * 16 + wfofs + 1];
    b1.f4[0] = Wq[(1 * 32 + lane) * 16 + wfofs];
    b1.f4[1] = Wq[(1 * 32 + lane) * 16 + wfofs + 1];

    for (int tk = 0; tk < 36; tk += 2) {
        // Chunk tk with b0; prefetch B(tk+2) into b0 behind it.
        conv_chunk(tk, b0, acc, sp, mrow, sel);
        {
            const int tkn = (tk + 2 < 36) ? tk + 2 : 34;
            b0.f4[0] = Wq[(tkn * 32 + lane) * 16 + wfofs];
            b0.f4[1] = Wq[(tkn * 32 + lane) * 16 + wfofs + 1];
        }
        // Chunk tk+1 with b1; prefetch B(tk+3) into b1 behind it.
        conv_chunk(tk + 1, b1, acc, sp, mrow, sel);
        {
            const int tkn = (tk + 3 < 36) ? tk + 3 : 35;
            b1.f4[0] = Wq[(tkn * 32 + lane) * 16 + wfofs];
            b1.f4[1] = Wq[(tkn * 32 + lane) * 16 + wfofs + 1];
        }
    }

    // Store: D layout -> VGPR v holds M = v + 8*(lane>=16), N = lane&15.
    const float cb = scal[8];
    const int co   = wv * 16 + (lane & 15);
    const int mofs = 8 * (lane >> 4);
#pragma unroll
    for (int g = 0; g < 2; g++) {
        float* orow = out + (size_t)(b * HH + (y0 + g)) * WW * COUT;
#pragma unroll
        for (int mt = 0; mt < 4; mt++) {
#pragma unroll
            for (int vg = 0; vg < 8; vg++) {
                int x = mt * 16 + vg + mofs;
                if (x < WW) orow[x * COUT + co] = acc[g * 4 + mt][vg] + cb;
            }
        }
    }
}

// ---------------- Host-side launcher --------------------------------------
extern "C" void kernel_launch(void* const* d_in, const int* in_sizes, int n_in,
                              void* d_out, int out_size, void* d_ws, size_t ws_size,
                              hipStream_t stream) {
    const float* X    = (const float*)d_in[0];
    const float* W    = (const float*)d_in[1];
    const float* beta = (const float*)d_in[2];
    const float* v    = (const float*)d_in[3];
    const float* bias = (const float*)d_in[4];
    const int N = in_sizes[2];   // 3

    char* ws = (char*)d_ws;
    _Float16* Ah  = (_Float16*)(ws);                 // 25,690,112 B activations (f16)
    _Float16* Wh  = (_Float16*)(ws + 25690112);      //    294,912 B W_eff (f16)
    float*    scal = (float*)(ws + 25985024);        // mean,std,alpha[5],sum,cbias

    k_stats<<<1, 256, 0, stream>>>(W, scal);
    k_alpha<<<1, 256, 0, stream>>>(W, beta, bias, N, scal);
    k_weff <<<(KW + 255) / 256, 256, 0, stream>>>(W, scal, Wh);

    const int n4 = NPIX * CIN / 4;                   // 3,211,264 float4 groups
    k_act  <<<(n4 + 255) / 256, 256, 0, stream>>>(X, beta, v, N, Ah, n4);

    k_conv <<<BATCH * (HH / 2), 256, 0, stream>>>(Ah, Wh, scal, (float*)d_out);
}
